// EvoformerBlock_64707977281602
// MI455X (gfx1250) — compile-verified, matrix operand order checked by reading, not run on hardware
//
#include <hip/hip_runtime.h>

// ---------------------------------------------------------------------------
// Evoformer block for gfx1250 (MI455X).
//  * All dense contractions: v_wmma_f32_16x16x32_f16 (f16 in, f32 acc).
//  * Activations & weights feeding GEMMs kept in f16 in global memory; GEMM
//    tiles staged with global_load_async_to_lds_b128 (ASYNCcnt), DOUBLE
//    BUFFERED: tile t+1's async DMA overlaps tile t's WMMA work.
//  * LN / softmax / gating / triangle einsum / OPM stay fp32 VALU.
//  * Running m and z states live in d_out (residual updates in place).
//  * Workspace ~460 MB of d_ws (9 f32 slots + 9 f16 slots + f16 weights).
//  * All call sites have K % 32 == 0 (kernel assumes it).
// ---------------------------------------------------------------------------

typedef __attribute__((ext_vector_type(16))) _Float16 v16h;
typedef __attribute__((ext_vector_type(8)))  _Float16 v8h;
typedef __attribute__((ext_vector_type(8)))  float    v8f;

#define TM 128
#define TN 64
#define TK 32
#define LDS_STRIDE 40     // halves per LDS row (80B, multiple of 16B)
#define USE_ASYNC_COPY 1

// ------------------------------ WMMA GEMM ----------------------------------
// D = epi( alpha * A @ B + bias ); A f16 [M,K]; B f16 [K,N] or [N,K] (transB).
// Outputs: C (f32, optional) and/or C16 (f16, optional), same layout.
__global__ __launch_bounds__(256) void k_gemm16(
    const _Float16* __restrict__ A, long long sAz, int lda,
    const _Float16* __restrict__ B, long long sBz, int ldb, int transB,
    float* __restrict__ C, _Float16* __restrict__ C16, long long sCz, int ldc,
    const float* __restrict__ bias, const float* __restrict__ rowmask,
    const float* __restrict__ mulg, const float* __restrict__ res,
    int M, int N, int K, float alpha, int act)
{
  A += (long long)blockIdx.z * sAz;
  B += (long long)blockIdx.z * sBz;
  const long long czoff = (long long)blockIdx.z * sCz;
  if (C)   C   += czoff;
  if (C16) C16 += czoff;

  __shared__ __align__(16) _Float16 As[2][TM * LDS_STRIDE];
  __shared__ __align__(16) _Float16 Bs[2][TN * LDS_STRIDE];
  __shared__ __align__(16) _Float16 dump_[8];   // sink for masked-off async chunks

  const int tid  = threadIdx.x;
  const int lane = tid & 31;
  const int wave = tid >> 5;
  const int wm   = wave >> 1;   // 0..3  (M direction)
  const int wn   = wave & 1;    // 0..1  (N direction)
  const int lh   = lane & 15;
  const int lg   = lane >> 4;
  const int m0   = blockIdx.y * TM;
  const int n0   = blockIdx.x * TN;

  // ---- K-invariant staging descriptors (validity hoisted out of loop) ----
  // A: 512 x 16B chunks, 2 per thread
  int ar[2], aseg[2]; long long agrow[2]; bool avalid[2];
#pragma unroll
  for (int t = 0; t < 2; ++t) {
    int cid = tid + t * 256;
    ar[t] = cid >> 2;  aseg[t] = cid & 3;
    int gm = m0 + ar[t];
    avalid[t] = (gm < M);
    agrow[t]  = (long long)(gm < M ? gm : (M - 1)) * lda;   // clamped
  }
  // B (trans): 256 x 16B chunks, 1 per thread
  int bn = tid >> 2, bseg = tid & 3;
  bool bvalid_t = (n0 + bn < N);
  long long bgrow_t = (long long)((n0 + bn < N) ? (n0 + bn) : (N - 1)) * ldb;
  // B (non-trans): vector load along n, transpose store
  int bk = tid >> 3, bnseg = tid & 7;
  int bgn_nt = n0 + bnseg * 8;
  bool bvalid_nt = (bgn_nt + 8 <= N);

  // ---- zero-fill chunks that are never written (both buffers, once) ------
#pragma unroll
  for (int bufi = 0; bufi < 2; ++bufi) {
    v8h z = {};
#pragma unroll
    for (int t = 0; t < 2; ++t)
      if (!avalid[t]) *(v8h*)&As[bufi][ar[t] * LDS_STRIDE + aseg[t] * 8] = z;
    if (transB) {
      if (!bvalid_t) *(v8h*)&Bs[bufi][bn * LDS_STRIDE + bseg * 8] = z;
    } else if (!bvalid_nt) {
#pragma unroll
      for (int j = 0; j < 8; ++j)
        Bs[bufi][(bnseg * 8 + j) * LDS_STRIDE + bk] = (_Float16)0.f;
    }
  }

  // ---- tile staging: fixed async-op count per thread (2 for A, +1 B) -----
  auto stage = [&](int buf, int kk) {
#pragma unroll
    for (int t = 0; t < 2; ++t) {
      _Float16* lp = avalid[t] ? &As[buf][ar[t] * LDS_STRIDE + aseg[t] * 8] : dump_;
      const _Float16* gp = A + agrow[t] + kk + aseg[t] * 8;
#if USE_ASYNC_COPY
      unsigned lds = (unsigned)(unsigned long long)(void*)lp;
      unsigned long long ga = (unsigned long long)gp;
      asm volatile("global_load_async_to_lds_b128 %0, %1, off"
                   :: "v"(lds), "v"(ga) : "memory");
#else
      *(v8h*)lp = *(const v8h*)gp;
#endif
    }
    if (transB) {
      _Float16* lp = bvalid_t ? &Bs[buf][bn * LDS_STRIDE + bseg * 8] : dump_;
      const _Float16* gp = B + bgrow_t + kk + bseg * 8;
#if USE_ASYNC_COPY
      unsigned lds = (unsigned)(unsigned long long)(void*)lp;
      unsigned long long ga = (unsigned long long)gp;
      asm volatile("global_load_async_to_lds_b128 %0, %1, off"
                   :: "v"(lds), "v"(ga) : "memory");
#else
      *(v8h*)lp = *(const v8h*)gp;
#endif
    } else if (bvalid_nt) {
      v8h val = *(const v8h*)(B + (long long)(kk + bk) * ldb + bgn_nt);
#pragma unroll
      for (int j = 0; j < 8; ++j)
        Bs[buf][(bnseg * 8 + j) * LDS_STRIDE + bk] = val[j];
    }
  };

  const int nk = K / TK;          // K % 32 == 0 at every call site
  v8f acc[2][2] = {};
  stage(0, 0);
  int cur = 0;

  for (int t = 0; t < nk; ++t) {
    const bool hasNext = (t + 1 < nk);
    if (hasNext) stage(cur ^ 1, (t + 1) * TK);   // overlap next DMA w/ compute
#if USE_ASYNC_COPY
    if (hasNext) {
      if (transB) asm volatile("s_wait_asynccnt 0x3" ::: "memory");
      else        asm volatile("s_wait_asynccnt 0x2" ::: "memory");
    } else {
      asm volatile("s_wait_asynccnt 0x0" ::: "memory");
    }
#endif
    __syncthreads();              // tile t resident in LDS for all waves

    v16h af[2], bf[2];
#pragma unroll
    for (int mi = 0; mi < 2; ++mi) {
      int row = wm * 32 + mi * 16 + lh;
      v8h lo = *reinterpret_cast<const v8h*>(&As[cur][row * LDS_STRIDE + lg * 8]);
      v8h hi = *reinterpret_cast<const v8h*>(&As[cur][row * LDS_STRIDE + lg * 8 + 16]);
#pragma unroll
      for (int i = 0; i < 8; ++i) { af[mi][i] = lo[i]; af[mi][i + 8] = hi[i]; }
    }
#pragma unroll
    for (int ni = 0; ni < 2; ++ni) {
      int col = wn * 32 + ni * 16 + lh;
      v8h lo = *reinterpret_cast<const v8h*>(&Bs[cur][col * LDS_STRIDE + lg * 8]);
      v8h hi = *reinterpret_cast<const v8h*>(&Bs[cur][col * LDS_STRIDE + lg * 8 + 16]);
#pragma unroll
      for (int i = 0; i < 8; ++i) { bf[ni][i] = lo[i]; bf[ni][i + 8] = hi[i]; }
    }
#pragma unroll
    for (int mi = 0; mi < 2; ++mi)
#pragma unroll
      for (int ni = 0; ni < 2; ++ni)
        acc[mi][ni] = __builtin_amdgcn_wmma_f32_16x16x32_f16(
            false, af[mi], false, bf[ni], (short)0, acc[mi][ni], false, false);
    __syncthreads();              // all waves done reading buf before restage
    cur ^= 1;
  }

  // ---- epilogue: vgpr r -> M = r + lg*8 ; lane lh -> N ---------------------
#pragma unroll
  for (int mi = 0; mi < 2; ++mi) {
#pragma unroll
    for (int ni = 0; ni < 2; ++ni) {
#pragma unroll
      for (int r = 0; r < 8; ++r) {
        int gm = m0 + wm * 32 + mi * 16 + lg * 8 + r;
        int gn = n0 + wn * 32 + ni * 16 + lh;
        if (gm < M && gn < N) {
          float v = acc[mi][ni][r] * alpha;
          if (bias)    v += bias[gn];
          if (act == 1) v = fmaxf(v, 0.f);
          else if (act == 2) v = 1.f / (1.f + expf(-v));
          if (rowmask) v *= rowmask[gm];
          if (mulg)    v *= mulg[(long long)gm * ldc + gn];
          if (res)     v += res[(long long)gm * ldc + gn];
          long long o = (long long)gm * ldc + gn;
          if (C)   C[o]   = v;
          if (C16) C16[o] = (_Float16)v;
        }
      }
    }
  }
}

// ------------------------------ LayerNorm ----------------------------------
__global__ void k_ln(const float* __restrict__ x, const float* __restrict__ g,
                     const float* __restrict__ b, float* __restrict__ y,
                     _Float16* __restrict__ y16, int C)
{
  long long row = blockIdx.x;
  const float* xr = x + row * C;
  int tid = threadIdx.x, wave = tid >> 5, lane = tid & 31;
  __shared__ float sw[4];
  float s = 0.f;
  for (int c = tid; c < C; c += 128) s += xr[c];
  for (int off = 16; off > 0; off >>= 1) s += __shfl_xor(s, off, 32);
  if (lane == 0) sw[wave] = s;
  __syncthreads();
  float mu = (sw[0] + sw[1] + sw[2] + sw[3]) / C;
  float v = 0.f;
  for (int c = tid; c < C; c += 128) { float d = xr[c] - mu; v += d * d; }
  for (int off = 16; off > 0; off >>= 1) v += __shfl_xor(v, off, 32);
  __syncthreads();
  if (lane == 0) sw[wave] = v;
  __syncthreads();
  float rstd = rsqrtf((sw[0] + sw[1] + sw[2] + sw[3]) / C + 1e-5f);
  for (int c = tid; c < C; c += 128) {
    float o = (xr[c] - mu) * rstd * g[c] + b[c];
    if (y)   y[row * C + c] = o;
    if (y16) y16[row * C + c] = (_Float16)o;
  }
}

// ------------------------------ softmax ------------------------------------
// reads f32 scores sc[h,q,k], adds mask/extra bias, writes f16 probs.
__global__ void k_softmax(const float* __restrict__ sc, _Float16* __restrict__ o16,
                          int Q, int Kk,
                          const float* __restrict__ mask, int maskStride,
                          const float* __restrict__ pb, int pbH)
{
  int q = blockIdx.x, h = blockIdx.z;
  long long rowoff = ((long long)h * Q + q) * Kk;
  int k = threadIdx.x, wave = k >> 5, lane = k & 31;
  __shared__ float s1[8], s2[8];
  float v = -3.0e38f;
  if (k < Kk) {
    v = sc[rowoff + k];
    if (mask) v += 1e9f * (mask[(long long)k * maskStride] - 1.f);
    if (pb)   v += pb[((long long)q * Kk + k) * pbH + h];
  }
  float m = v;
  for (int off = 16; off > 0; off >>= 1) m = fmaxf(m, __shfl_xor(m, off, 32));
  if (lane == 0) s1[wave] = m;
  __syncthreads();
  m = s1[0];
#pragma unroll
  for (int i = 1; i < 8; ++i) m = fmaxf(m, s1[i]);
  float e = (k < Kk) ? expf(v - m) : 0.f;
  float s = e;
  for (int off = 16; off > 0; off >>= 1) s += __shfl_xor(s, off, 32);
  if (lane == 0) s2[wave] = s;
  __syncthreads();
  s = s2[0] + s2[1] + s2[2] + s2[3] + s2[4] + s2[5] + s2[6] + s2[7];
  if (k < Kk) o16[rowoff + k] = (_Float16)(e / s);
}

// -------------------- small projection out[m,h] = x@w ----------------------
__global__ void k_small_nk(const float* __restrict__ x, const float* __restrict__ w,
                           float* __restrict__ out, long long M, int C, int H)
{
  long long idx = (long long)blockIdx.x * blockDim.x + threadIdx.x;
  if (idx >= M * H) return;
  int h = (int)(idx % H);
  long long m = idx / H;
  float acc = 0.f;
  for (int c = 0; c < C; ++c) acc += x[m * C + c] * w[c * H + h];
  out[m * H + h] = acc;
}

// ------------------- gated product: o16 = f16(a * g) -----------------------
__global__ void k_mul16(const float* __restrict__ a, const float* __restrict__ g,
                        _Float16* __restrict__ o16, long long n)
{
  long long idx = (long long)blockIdx.x * blockDim.x + threadIdx.x;
  if (idx < n) o16[idx] = (_Float16)(a[idx] * g[idx]);
}

// -------- weight transpose+convert: wt[n*K+k] = f16(w[k*N+n]) --------------
__global__ void k_cvt_t(const float* __restrict__ w, _Float16* __restrict__ wt,
                        int K, int N)
{
  long long idx = (long long)blockIdx.x * blockDim.x + threadIdx.x;
  if (idx >= (long long)K * N) return;
  int n = (int)(idx % N);
  int k = (int)(idx / N);
  wt[(long long)n * K + k] = (_Float16)w[idx];
}

// ---------------- transpose-01 (with optional accumulate) ------------------
// dst[D0,D1,C]; dst[i,j,c] (+)= src[j,i,c]  where src is [D1,D0,C]
__global__ void k_t01(float* __restrict__ dst, const float* __restrict__ src,
                      int D0, int D1, int C, int accumulate)
{
  long long idx = (long long)blockIdx.x * blockDim.x + threadIdx.x;
  long long total = (long long)D0 * D1 * C;
  if (idx >= total) return;
  int c = (int)(idx % C);
  long long t = idx / C;
  int j = (int)(t % D1);
  int i = (int)(t / D1);
  float v = src[((long long)j * D0 + i) * C + c];
  if (accumulate) dst[idx] += v; else dst[idx] = v;
}

// ----------------- outer-product-mean block per (i,j) ----------------------
__global__ void k_opm(const float* __restrict__ a, const float* __restrict__ bmat,
                      const float* __restrict__ wo, const float* __restrict__ bo,
                      const float* __restrict__ mask, float* __restrict__ z)
{
  int j = blockIdx.x, i = blockIdx.y;
  __shared__ float sa[128 * 32], sb[128 * 32], so_[1024], sn[4];
  int tid = threadIdx.x, wave = tid >> 5, lane = tid & 31;
  for (int idx = tid; idx < 128 * 32; idx += 128) {
    int s = idx >> 5, c = idx & 31;
    sa[idx] = a[((long long)s * 256 + i) * 32 + c];
    sb[idx] = bmat[((long long)s * 256 + j) * 32 + c];
  }
  float nm = mask[(long long)tid * 256 + i] * mask[(long long)tid * 256 + j];
  for (int off = 16; off > 0; off >>= 1) nm += __shfl_xor(nm, off, 32);
  if (lane == 0) sn[wave] = nm;
  __syncthreads();
  float norm = sn[0] + sn[1] + sn[2] + sn[3];
  for (int idx = tid; idx < 1024; idx += 128) {
    int c = idx >> 5, d = idx & 31;
    float accv = 0.f;
    for (int s = 0; s < 128; ++s) accv += sa[s * 32 + c] * sb[s * 32 + d];
    so_[idx] = accv;
  }
  __syncthreads();
  float accv = bo[tid];
  for (int cd = 0; cd < 1024; ++cd) accv += so_[cd] * wo[(long long)cd * 128 + tid];
  z[((long long)i * 256 + j) * 128 + tid] += accv / (1e-3f + norm);
}

// --------------- triangle multiplicative einsum ----------------------------
__global__ void k_trimul(const float* __restrict__ a, const float* __restrict__ b,
                         float* __restrict__ x, int outgoing)
{
  int j = blockIdx.x, i = blockIdx.y, c = threadIdx.x;
  float acc = 0.f;
  if (outgoing) {
    for (int k = 0; k < 256; ++k)
      acc += a[((long long)i * 256 + k) * 128 + c] * b[((long long)j * 256 + k) * 128 + c];
  } else {
    for (int k = 0; k < 256; ++k)
      acc += a[((long long)k * 256 + i) * 128 + c] * b[((long long)k * 256 + j) * 128 + c];
  }
  x[((long long)i * 256 + j) * 128 + c] = acc;
}

// ===========================================================================
//                                HOST SIDE
// ===========================================================================
static void launch_gemm(hipStream_t st,
                        const _Float16* A, long long sAz, int lda,
                        const _Float16* B, long long sBz, int ldb, int tB,
                        float* C, _Float16* C16, long long sCz, int ldc,
                        const float* bias, const float* rowmask,
                        const float* mulg, const float* res,
                        int M, int N, int K, float alpha, int act, int batch)
{
  dim3 grid((N + TN - 1) / TN, (M + TM - 1) / TM, batch);
  k_gemm16<<<grid, dim3(256), 0, st>>>(A, sAz, lda, B, sBz, ldb, tB,
                                       C, C16, sCz, ldc, bias, rowmask, mulg, res,
                                       M, N, K, alpha, act);
}

extern "C" void kernel_launch(void* const* d_in, const int* in_sizes, int n_in,
                              void* d_out, int out_size, void* d_ws, size_t ws_size,
                              hipStream_t stream)
{
  (void)in_sizes; (void)n_in; (void)out_size; (void)ws_size;
  constexpr int S_ = 128, R_ = 256;
  constexpr long long NM = (long long)S_ * R_;   // 32768 MSA tokens (C=256)
  constexpr long long NZ = (long long)R_ * R_;   // 65536 pair tokens (C=128)
  const float ALPHA = 0.17677669529663689f;      // 1/sqrt(32)

  auto P = [&](int i) { return (const float*)d_in[i]; };
  const float* msa_mask  = P(2);
  const float* pair_mask = P(3);

  float* out_m = (float*)d_out;                  // running m  [S,R,256]
  float* out_z = out_m + NM * 256;               // running z  [R,R,128]

  // ---------------- workspace layout ----------------
  const long long SLOT = 8388608;                // elements per slot
  float* W = (float*)d_ws;
  float* S0 = W;             float* S1 = W + 1 * SLOT; float* S2 = W + 2 * SLOT;
  float* S3 = W + 3 * SLOT;  float* S4 = W + 4 * SLOT; float* S5 = W + 5 * SLOT;
  float* S6 = W + 6 * SLOT;  float* S7 = W + 7 * SLOT; float* S8 = W + 8 * SLOT;
  float* X0 = W + 9 * SLOT;            // small bias proj (<=524288 f32)
  float* X1 = X0 + 524288;             // f32 score buffer (<=524288 f32)
  _Float16* HB = (_Float16*)(X1 + 524288);
  _Float16* H0 = HB;             _Float16* H2 = HB + 2 * SLOT;
  _Float16* H3 = HB + 3 * SLOT;  _Float16* H4 = HB + 4 * SLOT;
  _Float16* H6 = HB + 6 * SLOT;  _Float16* H8 = HB + 8 * SLOT;
  _Float16* X1h = HB + 9 * SLOT;       // f16 probs (<=524288)
  _Float16* WH  = X1h + 524288;        // f16 transposed weight pool

  // ---------------- weight convert+transpose (once per call) ---------------
  const _Float16* w16[97];
  {
    struct WSpec { int pi, K, N; };
    static const WSpec ws_[] = {
      {9,256,256},{10,256,256},{11,256,256},{12,256,256},{14,256,256},   // row_
      {18,256,256},{19,256,256},{20,256,256},{21,256,256},{23,256,256},  // col_
      {27,256,1024},{29,1024,256},                                       // mtr_
      {33,256,32},{35,256,32},                                           // opm_
      {41,128,128},{43,128,128},{45,128,128},{47,128,128},{49,128,128},{53,128,128}, // tmo_
      {57,128,128},{59,128,128},{61,128,128},{63,128,128},{65,128,128},{69,128,128}, // tmi_
      {74,128,128},{75,128,128},{76,128,128},{77,128,128},{79,128,128},  // tas_
      {84,128,128},{85,128,128},{86,128,128},{87,128,128},{89,128,128},  // tae_
      {93,128,512},{95,512,128},                                         // ptr_
    };
    long long off = 0;
    for (const auto& s : ws_) {
      long long n = (long long)s.K * s.N;
      w16[s.pi] = WH + off;
      k_cvt_t<<<(int)((n + 255) / 256), 256, 0, stream>>>(P(s.pi), WH + off, s.K, s.N);
      off += n;
    }
  }

  // init running states
  hipMemcpyAsync(out_m, d_in[0], sizeof(float) * NM * 256, hipMemcpyDeviceToDevice, stream);
  hipMemcpyAsync(out_z, d_in[1], sizeof(float) * NZ * 128, hipMemcpyDeviceToDevice, stream);

  // ---------------- MSA row attention with pair bias (row_: base 4) --------
  {
    k_ln<<<(int)NM, 128, 0, stream>>>(out_m, P(4), P(5), nullptr, H0, 256);  // m_ln (f16)
    k_ln<<<(int)NZ, 128, 0, stream>>>(out_z, P(6), P(7), S1, nullptr, 128);  // z_ln (f32)
    k_small_nk<<<(int)((NZ * 8 + 255) / 256), 256, 0, stream>>>(S1, P(8), X0, NZ, 128, 8);
    launch_gemm(stream, H0,0,256, w16[9] ,0,256,1, nullptr,H2, 0,256, nullptr,nullptr,nullptr,nullptr, NM,256,256, ALPHA,0,1); // q
    launch_gemm(stream, H0,0,256, w16[10],0,256,1, nullptr,H3, 0,256, nullptr,nullptr,nullptr,nullptr, NM,256,256, 1.f,0,1);   // k
    launch_gemm(stream, H0,0,256, w16[11],0,256,1, nullptr,H4, 0,256, nullptr,nullptr,nullptr,nullptr, NM,256,256, 1.f,0,1);   // v
    launch_gemm(stream, H0,0,256, w16[12],0,256,1, S5,nullptr, 0,256, P(13),nullptr,nullptr,nullptr,  NM,256,256, 1.f,2,1);   // gate
    for (int s = 0; s < S_; ++s) {
      const _Float16* Qs = H2 + (long long)s * R_ * 256;
      const _Float16* Ks = H3 + (long long)s * R_ * 256;
      const _Float16* Vs = H4 + (long long)s * R_ * 256;
      launch_gemm(stream, Qs,32,256, Ks,32,256,1, X1,nullptr,65536,256, nullptr,nullptr,nullptr,nullptr, 256,256,32, 1.f,0,8);
      k_softmax<<<dim3(256,1,8), 256, 0, stream>>>(X1, X1h, 256,256, msa_mask + (long long)s*R_, 1, X0, 8);
      launch_gemm(stream, X1h,65536,256, Vs,32,256,0, S6 + (long long)s*R_*256,nullptr, 32,256,
                  nullptr,nullptr,nullptr,nullptr, 256,32,256, 1.f,0,8);
    }
    k_mul16<<<(int)((NM*256 + 255)/256), 256, 0, stream>>>(S6, S5, H6, NM*256);
    launch_gemm(stream, H6,0,256, w16[14],0,256,1, out_m,nullptr,0,256, P(15),nullptr,nullptr,out_m, NM,256,256, 1.f,0,1);
  }

  // ---------------- MSA column attention (col_: base 16) -------------------
  {
    k_t01<<<(int)((NM*256 + 255)/256), 256, 0, stream>>>(S0, out_m, R_, S_, 256, 0);  // mt
    k_ln<<<(int)NM, 128, 0, stream>>>(S0, P(16), P(17), nullptr, H0, 256);
    launch_gemm(stream, H0,0,256, w16[18],0,256,1, nullptr,H2,0,256, nullptr,nullptr,nullptr,nullptr, NM,256,256, ALPHA,0,1);
    launch_gemm(stream, H0,0,256, w16[19],0,256,1, nullptr,H3,0,256, nullptr,nullptr,nullptr,nullptr, NM,256,256, 1.f,0,1);
    launch_gemm(stream, H0,0,256, w16[20],0,256,1, nullptr,H4,0,256, nullptr,nullptr,nullptr,nullptr, NM,256,256, 1.f,0,1);
    launch_gemm(stream, H0,0,256, w16[21],0,256,1, S5,nullptr,0,256, P(22),nullptr,nullptr,nullptr,  NM,256,256, 1.f,2,1);
    for (int i = 0; i < R_; ++i) {
      const _Float16* Qi = H2 + (long long)i * S_ * 256;
      const _Float16* Ki = H3 + (long long)i * S_ * 256;
      const _Float16* Vi = H4 + (long long)i * S_ * 256;
      launch_gemm(stream, Qi,32,256, Ki,32,256,1, X1,nullptr,16384,128, nullptr,nullptr,nullptr,nullptr, 128,128,32, 1.f,0,8);
      k_softmax<<<dim3(128,1,8), 256, 0, stream>>>(X1, X1h, 128,128, msa_mask + i, R_, nullptr, 0);
      launch_gemm(stream, X1h,16384,128, Vi,32,256,0, S6 + (long long)i*S_*256,nullptr, 32,256,
                  nullptr,nullptr,nullptr,nullptr, 128,32,128, 1.f,0,8);
    }
    k_mul16<<<(int)((NM*256 + 255)/256), 256, 0, stream>>>(S6, S5, H6, NM*256);
    launch_gemm(stream, H6,0,256, w16[23],0,256,1, S7,nullptr,0,256, P(24),nullptr,nullptr,nullptr, NM,256,256, 1.f,0,1);
    k_t01<<<(int)((NM*256 + 255)/256), 256, 0, stream>>>(out_m, S7, S_, R_, 256, 1);   // m += X^T
  }

  // ---------------- MSA transition (mtr_: base 25) -------------------------
  {
    k_ln<<<(int)NM, 128, 0, stream>>>(out_m, P(25), P(26), nullptr, H0, 256);
    for (long long c0 = 0; c0 < NM; c0 += 4096) {
      launch_gemm(stream, H0 + c0*256,0,256, w16[27],0,256,1, nullptr,H8,0,1024, P(28),nullptr,nullptr,nullptr,
                  4096,1024,256, 1.f,1,1);
      launch_gemm(stream, H8,0,1024, w16[29],0,1024,1, out_m + c0*256,nullptr,0,256, P(30), msa_mask + c0, nullptr,
                  out_m + c0*256, 4096,256,1024, 1.f,0,1);
    }
  }

  // ---------------- outer product mean (opm_: base 31) ---------------------
  {
    k_ln<<<(int)NM, 128, 0, stream>>>(out_m, P(31), P(32), nullptr, H0, 256);
    launch_gemm(stream, H0,0,256, w16[33],0,256,1, S2,nullptr,0,32, P(34), msa_mask, nullptr,nullptr, NM,32,256, 1.f,0,1);
    launch_gemm(stream, H0,0,256, w16[35],0,256,1, S3,nullptr,0,32, P(36), msa_mask, nullptr,nullptr, NM,32,256, 1.f,0,1);
    k_opm<<<dim3(R_, R_), 128, 0, stream>>>(S2, S3, P(37), P(38), msa_mask, out_z);
  }

  // ---------------- triangle multiplicative updates ------------------------
  auto tri_mul = [&](int b, int outgoing) {
    k_ln<<<(int)NZ, 128, 0, stream>>>(out_z, P(b+0), P(b+1), nullptr, H0, 128);  // zl (f16)
    launch_gemm(stream, H0,0,128, w16[b+4],0,128,1, S1,nullptr,0,128, P(b+5),nullptr,nullptr,nullptr, NZ,128,128, 1.f,2,1); // sig(ag)
    launch_gemm(stream, H0,0,128, w16[b+2],0,128,1, S2,nullptr,0,128, P(b+3), pair_mask, S1, nullptr, NZ,128,128, 1.f,0,1); // a
    launch_gemm(stream, H0,0,128, w16[b+8],0,128,1, S1,nullptr,0,128, P(b+9),nullptr,nullptr,nullptr, NZ,128,128, 1.f,2,1); // sig(bg)
    launch_gemm(stream, H0,0,128, w16[b+6],0,128,1, S3,nullptr,0,128, P(b+7), pair_mask, S1, nullptr, NZ,128,128, 1.f,0,1); // b
    k_trimul<<<dim3(R_, R_), 128, 0, stream>>>(S2, S3, S4, outgoing);
    k_ln<<<(int)NZ, 128, 0, stream>>>(S4, P(b+12), P(b+13), nullptr, H4, 128);   // ln_out -> f16
    launch_gemm(stream, H0,0,128, w16[b+10],0,128,1, S1,nullptr,0,128, P(b+11),nullptr,nullptr,nullptr, NZ,128,128, 1.f,2,1); // g
    launch_gemm(stream, H4,0,128, w16[b+14],0,128,1, out_z,nullptr,0,128, P(b+15), nullptr, S1, out_z, NZ,128,128, 1.f,0,1);
  };
  tri_mul(39, 1);   // tmo_  outgoing
  tri_mul(55, 0);   // tmi_  incoming

  // ---------------- triangle attention (tas_: 71 / tae_: 81) ---------------
  auto tri_attn = [&](int b, bool endNode) {
    const float* zin = out_z;
    if (endNode) {
      k_t01<<<(int)((NZ*128 + 255)/256), 256, 0, stream>>>(S7, out_z, R_, R_, 128, 0);
      zin = S7;
    }
    k_ln<<<(int)NZ, 128, 0, stream>>>(zin, P(b+0), P(b+1), S0, H0, 128);
    k_small_nk<<<(int)((NZ*4 + 255)/256), 256, 0, stream>>>(S0, P(b+2), X0, NZ, 128, 4);
    launch_gemm(stream, H0,0,128, w16[b+3],0,128,1, nullptr,H2,0,128, nullptr,nullptr,nullptr,nullptr, NZ,128,128, ALPHA,0,1);
    launch_gemm(stream, H0,0,128, w16[b+4],0,128,1, nullptr,H3,0,128, nullptr,nullptr,nullptr,nullptr, NZ,128,128, 1.f,0,1);
    launch_gemm(stream, H0,0,128, w16[b+5],0,128,1, nullptr,H4,0,128, nullptr,nullptr,nullptr,nullptr, NZ,128,128, 1.f,0,1);
    launch_gemm(stream, H0,0,128, w16[b+6],0,128,1, S5,nullptr,0,128, P(b+7),nullptr,nullptr,nullptr,  NZ,128,128, 1.f,2,1);
    for (int i = 0; i < R_; ++i) {
      const _Float16* Qi = H2 + (long long)i * R_ * 128;
      const _Float16* Ki = H3 + (long long)i * R_ * 128;
      const _Float16* Vi = H4 + (long long)i * R_ * 128;
      launch_gemm(stream, Qi,32,128, Ki,32,128,1, X1,nullptr,65536,256, nullptr,nullptr,nullptr,nullptr, 256,256,32, 1.f,0,4);
      const float* mrow = endNode ? (pair_mask + i) : (pair_mask + (long long)i * R_);
      int mstride = endNode ? R_ : 1;
      k_softmax<<<dim3(256,1,4), 256, 0, stream>>>(X1, X1h, 256,256, mrow, mstride, X0, 4);
      launch_gemm(stream, X1h,65536,256, Vi,32,128,0, S6 + (long long)i*R_*128,nullptr, 32,128,
                  nullptr,nullptr,nullptr,nullptr, 256,32,256, 1.f,0,4);
    }
    k_mul16<<<(int)((NZ*128 + 255)/256), 256, 0, stream>>>(S6, S5, H6, NZ*128);
    if (!endNode) {
      launch_gemm(stream, H6,0,128, w16[b+8],0,128,1, out_z,nullptr,0,128, P(b+9),nullptr,nullptr,out_z, NZ,128,128, 1.f,0,1);
    } else {
      launch_gemm(stream, H6,0,128, w16[b+8],0,128,1, S1,nullptr,0,128, P(b+9),nullptr,nullptr,nullptr, NZ,128,128, 1.f,0,1);
      k_t01<<<(int)((NZ*128 + 255)/256), 256, 0, stream>>>(out_z, S1, R_, R_, 128, 1);
    }
  };
  tri_attn(71, false);  // starting node
  tri_attn(81, true);   // ending node

  // ---------------- pair transition (ptr_: base 91) ------------------------
  {
    k_ln<<<(int)NZ, 128, 0, stream>>>(out_z, P(91), P(92), nullptr, H0, 128);
    for (long long c0 = 0; c0 < NZ; c0 += 8192) {
      launch_gemm(stream, H0 + c0*128,0,128, w16[93],0,128,1, nullptr,H8,0,512, P(94),nullptr,nullptr,nullptr,
                  8192,512,128, 1.f,1,1);
      launch_gemm(stream, H8,0,512, w16[95],0,512,1, out_z + c0*128,nullptr,0,128, P(96), pair_mask + c0, nullptr,
                  out_z + c0*128, 8192,128,512, 1.f,0,1);
    }
  }
}
